// GraphMoEDualRouter_10101763080598
// MI455X (gfx1250) — matrix operand order repaired and checked by reading, not compiled, not run on hardware
//
#include <hip/hip_runtime.h>
#include <math.h>

// Problem constants (fixed by the reference).
#define N_NODES 50000
#define N_EDGES 1600000
#define NGRAPH  64
#define HID     256
#define NEXP    8
#define NLAY    3

typedef __attribute__((ext_vector_type(16))) __bf16 v16bf;
typedef __attribute__((ext_vector_type(8)))  __bf16 v8bf;
typedef __attribute__((ext_vector_type(8)))  float  v8f;

// ---------- bf16 helpers (bit-exact, no header dependence) ----------
static __device__ __forceinline__ unsigned short f2bf(float f) {
  union { float f; unsigned int u; } v; v.f = f;
  unsigned int u = v.u;
  if ((u & 0x7fffffffu) > 0x7f800000u) return (unsigned short)((u >> 16) | 0x0040u); // quiet NaN
  unsigned int lsb = (u >> 16) & 1u;
  u += 0x7fffu + lsb;                       // round to nearest even
  return (unsigned short)(u >> 16);
}
static __device__ __forceinline__ float bf2f(unsigned short h) {
  union { unsigned int u; float f; } v; v.u = ((unsigned int)h) << 16; return v.f;
}

// ---------- tiny utility kernels ----------
__global__ void k_zero_u32(unsigned int* p, long long n) {
  long long i = (long long)blockIdx.x * blockDim.x + threadIdx.x;
  long long st = (long long)gridDim.x * blockDim.x;
  for (; i < n; i += st) p[i] = 0u;
}

__global__ void k_f32_to_bf16(const float* a, unsigned short* b, long long n) {
  long long i = (long long)blockIdx.x * blockDim.x + threadIdx.x;
  long long st = (long long)gridDim.x * blockDim.x;
  for (; i < n; i += st) b[i] = f2bf(a[i]);
}

// W: [nmat][K][O] f32 (row-major) -> WT: [nmat][O][K] bf16 (so a B-fragment
// column becomes a contiguous row -> single 32B load per lane in the GEMM).
__global__ void k_transpose_bf16(const float* W, unsigned short* WT, long long total, int K, int O) {
  long long i = (long long)blockIdx.x * blockDim.x + threadIdx.x;
  long long st = (long long)gridDim.x * blockDim.x;
  for (; i < total; i += st) {
    int o = (int)(i % O);
    long long t = i / O;
    int k = (int)(t % K);
    long long m = t / K;
    WT[(m * O + o) * (long long)K + k] = f2bf(W[i]);
  }
}

// ---------- graph size features ----------
__global__ void k_count_nodes(const int* batch, int* ncnt) {
  int n = blockIdx.x * blockDim.x + threadIdx.x;
  if (n < N_NODES) atomicAdd(&ncnt[batch[n]], 1);
}
__global__ void k_count_edges(const int* src, const int* batch, int* ecnt) {
  int e = blockIdx.x * blockDim.x + threadIdx.x;
  if (e < N_EDGES) atomicAdd(&ecnt[batch[src[e]]], 1);
}

// ---------- encoder: h = relu(x[:,4:10] @ W_enc + b_enc), bf16 out ----------
__global__ void k_encoder(const float* x, const float* Wenc, const float* benc,
                          unsigned short* hbf) {
  long long i = (long long)blockIdx.x * blockDim.x + threadIdx.x;
  if (i >= (long long)N_NODES * HID) return;
  int c = (int)(i & (HID - 1));
  long long n = i >> 8;
  const float* xr = x + n * 10 + 4;   // bus slice [4:10]
  float acc = benc[c];
  #pragma unroll
  for (int k = 0; k < 6; ++k) acc = fmaf(xr[k], Wenc[k * HID + c], acc);
  hbf[i] = f2bf(fmaxf(acc, 0.0f));
}

// ---------- neighbor aggregation: agg[dst] += z[src] (fp32 accumulate) ----------
struct __align__(16) BF8 { unsigned short h[8]; };
__global__ void k_scatter(const int* __restrict__ src, const int* __restrict__ dst,
                          const unsigned short* __restrict__ z, float* __restrict__ agg) {
  long long total = (long long)N_EDGES * 32;   // 32 groups of 8 features per edge
  long long i = (long long)blockIdx.x * blockDim.x + threadIdx.x;
  long long st = (long long)gridDim.x * blockDim.x;
  for (; i < total; i += st) {
    int e = (int)(i >> 5);
    int g = (int)(i & 31) * 8;
    int s = src[e], d = dst[e];
    BF8 v = *reinterpret_cast<const BF8*>(z + (size_t)s * HID + g);
    float* ap = agg + (size_t)d * HID + g;
    #pragma unroll
    for (int j = 0; j < 8; ++j) atomicAdd(ap + j, bf2f(v.h[j]));
  }
}

// ---------- WMMA fragment loads (wave32; contiguous k order, not the ISA table order) ----------
// A 16x32 bf16: lanes 0-15 = rows M=0..15 carrying K = kb+0..15;
//               lanes 16-31 = same rows carrying K = kb+16..31.
static __device__ __forceinline__ v16bf loadA(const unsigned short* row, int kb, int half) {
  return *reinterpret_cast<const v16bf*>(row + kb + half * 16);
}
// B 32x16 bf16 from pre-transposed weights WT[col][k]: lanes 0-15 = col N=lane,
// K = kb+0..15; lanes 16-31 = col N=lane-16, K = kb+16..31 (contiguous 32B).
static __device__ __forceinline__ v16bf loadB(const unsigned short* row, int kb, int half) {
  return *reinterpret_cast<const v16bf*>(row + kb + half * 16);
}

// ---------- main GEMM: C = A1@B1 (+ A2@B2) + bias, with 3 epilogues ----------
// MODE 0: two-operand GraphConv layer, ReLU, bf16 out   (hidden layers)
// MODE 1: single-operand router GEMM,   ReLU, bf16 out  (relu(h@Ws1+bs1))
// MODE 2: two-operand final layer, no ReLU, fused MoE combine:
//         out[n][c] += sparse[n][e] * (acc + bias[c])   (fp32, no atomics)
template <int MODE>
__global__ void __launch_bounds__(256)
k_gemm(const unsigned short* __restrict__ A1, const unsigned short* __restrict__ A2,
       const unsigned short* __restrict__ B1T, const unsigned short* __restrict__ B2T,
       const float* __restrict__ bias, const float* __restrict__ sparse, int expert,
       unsigned short* __restrict__ Obf, float* __restrict__ Of32, int Nrows) {
  const int wave = (blockIdx.x * blockDim.x + threadIdx.x) >> 5;
  const int lane = threadIdx.x & 31;
  const int rowTiles = Nrows >> 4;            // 3125, exact
  const int rt = wave % rowTiles;             // adjacent waves share the B tile -> L2 reuse
  const int ct = wave / rowTiles;
  if (ct >= HID / 16) return;                 // wave-uniform (EXEC stays all-ones for WMMA)
  const int half = lane >> 4, m = lane & 15;
  const int rowbase = rt * 16, colbase = ct * 16;

  v8f c = {};
  {
    const unsigned short* ar = A1 + (size_t)(rowbase + m) * HID;
    const unsigned short* br = B1T + (size_t)(colbase + m) * HID;
    #pragma unroll
    for (int kb = 0; kb < HID; kb += 32) {
      v16bf a = loadA(ar, kb, half);
      v16bf b = loadB(br, kb, half);
      c = __builtin_amdgcn_wmma_f32_16x16x32_bf16(false, a, false, b, (short)0, c, false, false);
    }
  }
  if (MODE != 1) {
    const unsigned short* ar = A2 + (size_t)(rowbase + m) * HID;
    const unsigned short* br = B2T + (size_t)(colbase + m) * HID;
    #pragma unroll
    for (int kb = 0; kb < HID; kb += 32) {
      v16bf a = loadA(ar, kb, half);
      v16bf b = loadB(br, kb, half);
      c = __builtin_amdgcn_wmma_f32_16x16x32_bf16(false, a, false, b, (short)0, c, false, false);
    }
  }

  // C/D layout: lane l, VGPR r -> M = r + 8*(l>=16), N = l&15.
  const int col = colbase + m;
  const float bv = bias[col];
  #pragma unroll
  for (int r = 0; r < 8; ++r) {
    int row = rowbase + r + half * 8;
    float v = c[r] + bv;
    if (MODE == 2) {
      float w = sparse[(size_t)row * NEXP + expert];
      size_t idx = (size_t)row * HID + col;
      Of32[idx] += w * v;                     // each (row,col) written by exactly one lane
    } else {
      Obf[(size_t)row * HID + col] = f2bf(fmaxf(v, 0.0f));
    }
  }
}

// ---------- second router stage: sem = r1 @ Ws2 + bs2 (one wave per node) ----------
__global__ void k_router_sem(const unsigned short* __restrict__ r1, const float* __restrict__ Ws2,
                             const float* __restrict__ bs2, float* __restrict__ sem) {
  int gt = blockIdx.x * blockDim.x + threadIdx.x;
  int node = gt >> 5, lane = gt & 31;
  if (node >= N_NODES) return;
  const unsigned short* row = r1 + (size_t)node * HID;
  float acc[NEXP] = {};
  for (int k = lane; k < HID; k += 32) {
    float t = bf2f(row[k]);
    const float* w = Ws2 + k * NEXP;
    #pragma unroll
    for (int j = 0; j < NEXP; ++j) acc[j] = fmaf(t, w[j], acc[j]);
  }
  #pragma unroll
  for (int j = 0; j < NEXP; ++j)
    for (int off = 16; off > 0; off >>= 1) acc[j] += __shfl_xor(acc[j], off, 32);
  if (lane == 0) {
    #pragma unroll
    for (int j = 0; j < NEXP; ++j) sem[(size_t)node * NEXP + j] = acc[j] + bs2[j];
  }
}

// ---------- structural router: only G=64 distinct rows (one wave per graph) ----------
__global__ void k_stru(const int* ncnt, const int* ecnt, const float* Wt1, const float* bt1,
                       const float* Wt2, const float* bt2, float* stru) {
  int gt = blockIdx.x * blockDim.x + threadIdx.x;
  int g = gt >> 5, lane = gt & 31;
  if (g >= NGRAPH) return;
  float s0 = log1pf((float)ncnt[g]);
  float s1 = log1pf((float)ecnt[g]);
  float acc[NEXP] = {};
  for (int k = lane; k < HID; k += 32) {
    float t = fmaxf(fmaf(s0, Wt1[k], fmaf(s1, Wt1[HID + k], bt1[k])), 0.0f);
    const float* w = Wt2 + k * NEXP;
    #pragma unroll
    for (int j = 0; j < NEXP; ++j) acc[j] = fmaf(t, w[j], acc[j]);
  }
  #pragma unroll
  for (int j = 0; j < NEXP; ++j)
    for (int off = 16; off > 0; off >>= 1) acc[j] += __shfl_xor(acc[j], off, 32);
  if (lane == 0) {
    #pragma unroll
    for (int j = 0; j < NEXP; ++j) stru[g * NEXP + j] = acc[j] + bt2[j];
  }
}

// ---------- softmax -> top-2 -> renormalize (softmax denom cancels) ----------
__global__ void k_route_finalize(const float* sem, const float* stru, const int* batch,
                                 float* sparse) {
  int n = blockIdx.x * blockDim.x + threadIdx.x;
  if (n >= N_NODES) return;
  int g = batch[n];
  float l[NEXP], mx = -3.0e38f;
  #pragma unroll
  for (int j = 0; j < NEXP; ++j) {
    l[j] = 0.5f * (sem[(size_t)n * NEXP + j] + stru[g * NEXP + j]);
    mx = fmaxf(mx, l[j]);
  }
  float p[NEXP];
  #pragma unroll
  for (int j = 0; j < NEXP; ++j) p[j] = expf(l[j] - mx);
  int i1 = 0;
  #pragma unroll
  for (int j = 1; j < NEXP; ++j) if (p[j] > p[i1]) i1 = j;   // first max (top_k tiebreak)
  int i2 = (i1 == 0) ? 1 : 0;
  #pragma unroll
  for (int j = 0; j < NEXP; ++j) if (j != i1 && p[j] > p[i2]) i2 = j;
  float inv = 1.0f / (p[i1] + p[i2]);
  #pragma unroll
  for (int j = 0; j < NEXP; ++j)
    sparse[(size_t)n * NEXP + j] = (j == i1 || j == i2) ? p[j] * inv : 0.0f;
}

// =====================================================================
extern "C" void kernel_launch(void* const* d_in, const int* in_sizes, int n_in,
                              void* d_out, int out_size, void* d_ws, size_t ws_size,
                              hipStream_t stream) {
  (void)in_sizes; (void)n_in; (void)out_size; (void)ws_size;
  const float* x       = (const float*)d_in[0];
  const int*   eidx    = (const int*)d_in[1];
  const int*   batch   = (const int*)d_in[2];
  const float* W_enc   = (const float*)d_in[3];
  const float* b_enc   = (const float*)d_in[4];
  const float* Ws1     = (const float*)d_in[5];
  const float* bs1     = (const float*)d_in[6];
  const float* Ws2     = (const float*)d_in[7];
  const float* bs2     = (const float*)d_in[8];
  const float* Wt1     = (const float*)d_in[9];
  const float* bt1     = (const float*)d_in[10];
  const float* Wt2     = (const float*)d_in[11];
  const float* bt2     = (const float*)d_in[12];
  const float* W_self  = (const float*)d_in[13];
  const float* W_neigh = (const float*)d_in[14];
  const float* b_exp   = (const float*)d_in[15];
  const int* src = eidx;
  const int* dst = eidx + N_EDGES;
  float* out = (float*)d_out;

  // ---- workspace carve (~190 MB total, 256B aligned) ----
  char* ws = (char*)d_ws; size_t off = 0;
  auto carve = [&](size_t bytes) -> char* {
    char* p = ws + off; off += (bytes + 255) & ~(size_t)255; return p;
  };
  unsigned short* hbf    = (unsigned short*)carve((size_t)N_NODES * HID * 2);
  unsigned short* zping  = (unsigned short*)carve((size_t)N_NODES * HID * 2);
  unsigned short* zpong  = (unsigned short*)carve((size_t)N_NODES * HID * 2);
  float*          aggf   = (float*)carve((size_t)N_NODES * HID * 4);
  unsigned short* aggbf  = (unsigned short*)carve((size_t)N_NODES * HID * 2);
  unsigned short* agg0bf = (unsigned short*)carve((size_t)N_NODES * HID * 2);
  unsigned short* WsT    = (unsigned short*)carve((size_t)NEXP * NLAY * HID * HID * 2);
  unsigned short* WnT    = (unsigned short*)carve((size_t)NEXP * NLAY * HID * HID * 2);
  unsigned short* Ws1T   = (unsigned short*)carve((size_t)HID * HID * 2);
  float* semb   = (float*)carve((size_t)N_NODES * NEXP * 4);
  float* sparse = (float*)carve((size_t)N_NODES * NEXP * 4);
  float* strub  = (float*)carve((size_t)NGRAPH * NEXP * 4);
  int* ncnt = (int*)carve(NGRAPH * 4);
  int* ecnt = (int*)carve(NGRAPH * 4);

  const int BS = 256;
  auto nb = [](long long n) { return (unsigned)((n + 255) / 256); };
  const long long NH = (long long)N_NODES * HID;
  const unsigned gemmBlocks = N_NODES / 8;   // (N/16 row tiles * 16 col tiles) waves / 8 per block

  // ---- setup ----
  k_zero_u32<<<nb(NH), BS, 0, stream>>>((unsigned*)out, NH);
  k_zero_u32<<<1, BS, 0, stream>>>((unsigned*)ncnt, NGRAPH);
  k_zero_u32<<<1, BS, 0, stream>>>((unsigned*)ecnt, NGRAPH);
  k_count_nodes<<<nb(N_NODES), BS, 0, stream>>>(batch, ncnt);
  k_count_edges<<<nb(N_EDGES), BS, 0, stream>>>(src, batch, ecnt);
  k_encoder<<<nb(NH), BS, 0, stream>>>(x, W_enc, b_enc, hbf);
  k_transpose_bf16<<<nb((long long)HID * HID), BS, 0, stream>>>(Ws1, Ws1T, (long long)HID * HID, HID, HID);
  k_transpose_bf16<<<nb((long long)NEXP * NLAY * HID * HID), BS, 0, stream>>>(
      W_self, WsT, (long long)NEXP * NLAY * HID * HID, HID, HID);
  k_transpose_bf16<<<nb((long long)NEXP * NLAY * HID * HID), BS, 0, stream>>>(
      W_neigh, WnT, (long long)NEXP * NLAY * HID * HID, HID, HID);

  // ---- routing ----
  k_gemm<1><<<gemmBlocks, BS, 0, stream>>>(hbf, nullptr, Ws1T, nullptr, bs1,
                                           nullptr, 0, zping, nullptr, N_NODES); // r1 in zping
  k_router_sem<<<nb((long long)N_NODES * 32), BS, 0, stream>>>(zping, Ws2, bs2, semb);
  k_stru<<<(NGRAPH * 32 + BS - 1) / BS, BS, 0, stream>>>(ncnt, ecnt, Wt1, bt1, Wt2, bt2, strub);
  k_route_finalize<<<nb(N_NODES), BS, 0, stream>>>(semb, strub, batch, sparse);

  // ---- layer-0 aggregation: z0 == h for every expert -> compute ONCE ----
  k_zero_u32<<<nb(NH), BS, 0, stream>>>((unsigned*)aggf, NH);
  k_scatter<<<nb((long long)N_EDGES * 32), BS, 0, stream>>>(src, dst, hbf, aggf);
  k_f32_to_bf16<<<nb(NH), BS, 0, stream>>>(aggf, agg0bf, NH);

  // ---- experts, sequential (z ping/pong reused; final layer fuses MoE combine) ----
  for (int e = 0; e < NEXP; ++e) {
    const size_t wstep = (size_t)HID * HID;
    const unsigned short* Ws_l0 = WsT + (size_t)(e * NLAY + 0) * wstep;
    const unsigned short* Ws_l1 = WsT + (size_t)(e * NLAY + 1) * wstep;
    const unsigned short* Ws_l2 = WsT + (size_t)(e * NLAY + 2) * wstep;
    const unsigned short* Wn_l0 = WnT + (size_t)(e * NLAY + 0) * wstep;
    const unsigned short* Wn_l1 = WnT + (size_t)(e * NLAY + 1) * wstep;
    const unsigned short* Wn_l2 = WnT + (size_t)(e * NLAY + 2) * wstep;
    const float* b0 = b_exp + (size_t)(e * NLAY + 0) * HID;
    const float* b1 = b_exp + (size_t)(e * NLAY + 1) * HID;
    const float* b2 = b_exp + (size_t)(e * NLAY + 2) * HID;

    // layer 0: z1 = relu(h@Ws0 + agg0@Wn0 + b0)
    k_gemm<0><<<gemmBlocks, BS, 0, stream>>>(hbf, agg0bf, Ws_l0, Wn_l0, b0,
                                             nullptr, 0, zping, nullptr, N_NODES);
    // layer 1
    k_zero_u32<<<nb(NH), BS, 0, stream>>>((unsigned*)aggf, NH);
    k_scatter<<<nb((long long)N_EDGES * 32), BS, 0, stream>>>(src, dst, zping, aggf);
    k_f32_to_bf16<<<nb(NH), BS, 0, stream>>>(aggf, aggbf, NH);
    k_gemm<0><<<gemmBlocks, BS, 0, stream>>>(zping, aggbf, Ws_l1, Wn_l1, b1,
                                             nullptr, 0, zpong, nullptr, N_NODES);
    // layer 2 + fused MoE combine into out (fp32, no relu)
    k_zero_u32<<<nb(NH), BS, 0, stream>>>((unsigned*)aggf, NH);
    k_scatter<<<nb((long long)N_EDGES * 32), BS, 0, stream>>>(src, dst, zpong, aggf);
    k_f32_to_bf16<<<nb(NH), BS, 0, stream>>>(aggf, aggbf, NH);
    k_gemm<2><<<gemmBlocks, BS, 0, stream>>>(zpong, aggbf, Ws_l2, Wn_l2, b2,
                                             sparse, e, nullptr, out, N_NODES);
  }
}
